// RowParallelTritonFusedMoeLinear_76647986365156
// MI455X (gfx1250) — matrix operand order repaired and checked by reading, not compiled, not run on hardware
//
#include <hip/hip_runtime.h>
#include <stdint.h>

#define T_TOK 8192
#define DIN   4096
#define DOUT  2048
#define NEXP  8

#define BM 128
#define BN 128
#define BK 32
#define NKT (DIN / BK)            // 128 k-steps
#define LDROW 80                  // padded LDS row stride in bytes (40 bf16)
#define ATILE (BM * LDROW)        // 10240 B
#define BTILE (BN * LDROW)        // 10240 B

typedef __attribute__((ext_vector_type(16))) __bf16 v16bf;
typedef __attribute__((ext_vector_type(2)))  __bf16 v2bf;
typedef __attribute__((ext_vector_type(8)))  float  v8f;
typedef __attribute__((ext_vector_type(4)))  float  vf4;
typedef __attribute__((ext_vector_type(4)))  unsigned int v4u;
typedef __attribute__((ext_vector_type(2)))  unsigned int v2u;

union Frag { v16bf h; v4u u[2]; };

// round-to-nearest-even f32 -> bf16, pack two into a dword
__device__ __forceinline__ unsigned int packbf2(float lo, float hi) {
#if defined(__has_builtin) && __has_builtin(__builtin_amdgcn_cvt_pk_bf16_f32)
    // single v_cvt_pk_bf16_f32 (RNE): D[15:0]=bf16(S0), D[31:16]=bf16(S1)
    v2bf v = __builtin_amdgcn_cvt_pk_bf16_f32(lo, hi);
    return __builtin_bit_cast(unsigned int, v);
#else
    // builtin is feature-gated off for gfx1250 in this clang; the CDNA5
    // instruction itself exists (since gfx950) -> issue it via inline asm.
    unsigned int r;
    asm("v_cvt_pk_bf16_f32 %0, %1, %2" : "=v"(r) : "v"(lo), "v"(hi));
    return r;
#endif
}

// ---------------- routing ----------------
__global__ void zero_meta(int* meta) {
    if (threadIdx.x < 24) meta[threadIdx.x] = 0;
}
__global__ void hist_kernel(const int* __restrict__ eid, int* __restrict__ meta) {
    int t = blockIdx.x * blockDim.x + threadIdx.x;
    if (t < T_TOK) atomicAdd(&meta[eid[t]], 1);
}
__global__ void scan_kernel(int* meta) {
    if (threadIdx.x == 0) {
        int off = 0;
        for (int e = 0; e < NEXP; ++e) {
            int c = meta[e];
            meta[8 + e]  = off;   // segment offset
            meta[16 + e] = off;   // scatter cursor
            off += c;
        }
    }
}
__global__ void scatter_kernel(const int* __restrict__ eid, int* __restrict__ meta,
                               int* __restrict__ perm) {
    int t = blockIdx.x * blockDim.x + threadIdx.x;
    if (t < T_TOK) {
        int p = atomicAdd(&meta[16 + eid[t]], 1);
        perm[p] = t;
    }
}

// ---------------- grouped GEMM ----------------
__global__ __launch_bounds__(256)
void moe_gemm_kernel(const float* __restrict__ x, const float* __restrict__ w,
                     const float* __restrict__ bias, const int* __restrict__ meta,
                     const int* __restrict__ perm, float* __restrict__ y) {
    const int e   = blockIdx.z;
    const int cnt = meta[e];
    const int off = meta[8 + e];
    const int mt  = blockIdx.y;
    if (mt * BM >= cnt) return;
    const int n0 = blockIdx.x * BN;

    __shared__ unsigned char lds[2 * (ATILE + BTILE)];

    const int tid = threadIdx.x;

    // ---- A staging assignment: 128 rows x 32 k; thread -> (row, 16-float k chunk)
    const int ar  = tid & 127;
    const int akh = (tid >> 7) * 16;           // 0 or 16 (k offset in floats)
    int mrow = mt * BM + ar;
    int tokA = perm[off + (mrow < cnt ? mrow : 0)];
    const float* aSrc = x + (size_t)tokA * DIN + akh;

    // ---- B staging assignment: 32 k rows x 128 n; thread -> (4 k rows, 4 n cols)
    const int bkq = tid >> 5;                  // 0..7 -> k rows bkq*4..+3
    const int bnb = (tid & 31) * 4;            // n offset 0..124
    const float* bSrc = w + (size_t)e * DIN * DOUT + (size_t)(bkq * 4) * DOUT + n0 + bnb;

    vf4 aR[4], bR[4];
    {   // prefetch tile 0
        const vf4* ap = (const vf4*)aSrc;
        aR[0] = ap[0]; aR[1] = ap[1]; aR[2] = ap[2]; aR[3] = ap[3];
#pragma unroll
        for (int r = 0; r < 4; ++r)
            bR[r] = *(const vf4*)(bSrc + (size_t)r * DOUT);
    }

    auto stage = [&](int buf) {
        unsigned char* A = lds + buf * (ATILE + BTILE);
        unsigned char* B = A + ATILE;
        // A: 16 f32 -> 16 bf16, linear in k
        v4u pa0 = { packbf2(aR[0][0], aR[0][1]), packbf2(aR[0][2], aR[0][3]),
                    packbf2(aR[1][0], aR[1][1]), packbf2(aR[1][2], aR[1][3]) };
        v4u pa1 = { packbf2(aR[2][0], aR[2][1]), packbf2(aR[2][2], aR[2][3]),
                    packbf2(aR[3][0], aR[3][1]), packbf2(aR[3][2], aR[3][3]) };
        v4u* dA = (v4u*)(A + ar * LDROW + akh * 2);
        dA[0] = pa0;
        dA[1] = pa1;
        // B: transpose to [n][k] during staging (4 k-vals per n -> b64 store)
#pragma unroll
        for (int c = 0; c < 4; ++c) {
            v2u d = { packbf2(bR[0][c], bR[1][c]), packbf2(bR[2][c], bR[3][c]) };
            *(v2u*)(B + (bnb + c) * LDROW + bkq * 8) = d;
        }
    };

    // ---- wave/tile coords: 8 waves -> 4 (m) x 2 (n); wave tile 32x64
    const int lane   = tid & 31;
    const int wid    = tid >> 5;
    const int wm     = wid & 3;
    const int wn     = wid >> 2;
    const int laneLo = lane & 15;
    const int laneHi = lane >> 4;

    v8f zero8 = {0.f, 0.f, 0.f, 0.f, 0.f, 0.f, 0.f, 0.f};
    v8f acc[2][4];
#pragma unroll
    for (int i = 0; i < 2; ++i)
#pragma unroll
        for (int j = 0; j < 4; ++j) acc[i][j] = zero8;

    stage(0);
    __syncthreads();

    int buf = 0;
    for (int kt = 0; kt < NKT; ++kt) {
        if (kt + 1 < NKT) {   // register prefetch of next tile (overlaps WMMA)
            const vf4* ap = (const vf4*)(aSrc + (size_t)(kt + 1) * BK);
            aR[0] = ap[0]; aR[1] = ap[1]; aR[2] = ap[2]; aR[3] = ap[3];
            const float* bp = bSrc + (size_t)(kt + 1) * BK * DOUT;
#pragma unroll
            for (int r = 0; r < 4; ++r)
                bR[r] = *(const vf4*)(bp + (size_t)r * DOUT);
        }

        const unsigned char* A = lds + buf * (ATILE + BTILE);
        const unsigned char* B = A + ATILE;
        Frag fa[2], fb[4];
        // A frag: lanes 0-15 take K 0..7 & 16..23, lanes 16-31 take K 8..15 & 24..31
#pragma unroll
        for (int i = 0; i < 2; ++i) {
            const unsigned char* p = A + (wm * 32 + i * 16 + laneLo) * LDROW + laneHi * 16;
            fa[i].u[0] = *(const v4u*)p;
            fa[i].u[1] = *(const v4u*)(p + 32);
        }
        // B frag: lanes 0-15 take K 0..7 & 16..23, lanes 16-31 take K 8..15 & 24..31
#pragma unroll
        for (int j = 0; j < 4; ++j) {
            const unsigned char* p = B + (wn * 64 + j * 16 + laneLo) * LDROW + laneHi * 16;
            fb[j].u[0] = *(const v4u*)p;
            fb[j].u[1] = *(const v4u*)(p + 32);
        }
#pragma unroll
        for (int i = 0; i < 2; ++i)
#pragma unroll
            for (int j = 0; j < 4; ++j)
                acc[i][j] = __builtin_amdgcn_wmma_f32_16x16x32_bf16(
                    false, fa[i].h, false, fb[j].h, (short)0, acc[i][j], false, false);

        if (kt + 1 < NKT) stage(buf ^ 1);   // writes other buffer: no conflict with reads
        __syncthreads();                    // single barrier/iter protects both hazards
        buf ^= 1;
    }

    // ---- epilogue: bias + scatter store (C/D layout: VGPR r -> M=r | M=8+r)
    int toks[2][8];
#pragma unroll
    for (int i = 0; i < 2; ++i)
#pragma unroll
        for (int r = 0; r < 8; ++r) {
            int m = mt * BM + wm * 32 + i * 16 + laneHi * 8 + r;
            toks[i][r] = (m < cnt) ? perm[off + m] : -1;
        }
#pragma unroll
    for (int j = 0; j < 4; ++j) {
        int n = n0 + wn * 64 + j * 16 + laneLo;
        float bv = bias[e * DOUT + n];
#pragma unroll
        for (int i = 0; i < 2; ++i)
#pragma unroll
            for (int r = 0; r < 8; ++r)
                if (toks[i][r] >= 0)
                    y[(size_t)toks[i][r] * DOUT + n] = acc[i][j][r] + bv;
    }
}

extern "C" void kernel_launch(void* const* d_in, const int* in_sizes, int n_in,
                              void* d_out, int out_size, void* d_ws, size_t ws_size,
                              hipStream_t stream) {
    (void)in_sizes; (void)n_in; (void)out_size; (void)ws_size;
    const float* x   = (const float*)d_in[0];
    const float* w   = (const float*)d_in[1];
    const float* b   = (const float*)d_in[2];
    const int*   eid = (const int*)d_in[3];
    float*       y   = (float*)d_out;

    int* meta = (int*)d_ws;        // [0..7] counts, [8..15] offsets, [16..23] cursors
    int* perm = meta + 24;         // [T_TOK]

    zero_meta<<<1, 32, 0, stream>>>(meta);
    hist_kernel<<<T_TOK / 256, 256, 0, stream>>>(eid, meta);
    scan_kernel<<<1, 1, 0, stream>>>(meta);
    scatter_kernel<<<T_TOK / 256, 256, 0, stream>>>(eid, meta, perm);
    moe_gemm_kernel<<<dim3(DOUT / BN, T_TOK / BM, NEXP), 256, 0, stream>>>(
        x, w, b, meta, perm, y);
}
